// BandProcessorWithHistory_56762287784317
// MI455X (gfx1250) — matrix operand order repaired and checked by reading, not compiled, not run on hardware
//
#include <hip/hip_runtime.h>
#include <hip/hip_bf16.h>
#include <math.h>

// ---------------------------------------------------------------------------
// BandProcessorWithHistory for MI455X (gfx1250, wave32, WMMA + TDM)
// B=8 T=2048 H=64 D=512.
// bf16 activations/weights in HBM (converted once), f32 accumulation in WMMA,
// TDM double-buffered LDS staging for all GEMMs.
// ---------------------------------------------------------------------------

typedef __attribute__((ext_vector_type(16))) __bf16 v16bf;
typedef __attribute__((ext_vector_type(8)))  __bf16 bf16x8;
typedef __attribute__((ext_vector_type(4)))  __bf16 bf16x4;
typedef __attribute__((ext_vector_type(8)))  float  v8f;
typedef __attribute__((ext_vector_type(4)))  unsigned int u32x4;
typedef __attribute__((ext_vector_type(8)))  int i32x8;
typedef __attribute__((ext_vector_type(4)))  int i32x4;

#define BP_B 8
#define BP_T 2048
#define BP_H 64
#define BP_D 512

// GEMM tiling
#define TM 128            // block tile M
#define TN 128            // block tile N
#define TK 64             // K slab per LDS stage
#define RS 72             // LDS row stride in elements (64 + 8 pad -> 144B rows)

#if defined(__HIP_DEVICE_COMPILE__) && __has_builtin(__builtin_amdgcn_tensor_load_to_lds)
#define HAVE_TDM 1
#else
#define HAVE_TDM 0
#endif

// ---------------------------------------------------------------------------
// fp32 -> bf16 conversion (n must be a multiple of 4)
// ---------------------------------------------------------------------------
__global__ __launch_bounds__(256)
void cvt_kernel(const float* __restrict__ x, __bf16* __restrict__ y, int n4)
{
    const int i = blockIdx.x * 256 + threadIdx.x;
    if (i >= n4) return;
    const float4 v = ((const float4*)x)[i];
    bf16x4 o;
    o[0] = (__bf16)v.x; o[1] = (__bf16)v.y; o[2] = (__bf16)v.z; o[3] = (__bf16)v.w;
    ((bf16x4*)y)[i] = o;
}

// ---------------------------------------------------------------------------
// LayerNorm (fp32 in, bf16 out): one 256-thread block per 512-float row.
// ---------------------------------------------------------------------------
__global__ __launch_bounds__(256)
void ln_kernel(const float* __restrict__ x, const float* __restrict__ g,
               const float* __restrict__ b, __bf16* __restrict__ y)
{
    __shared__ float red[8];
    const int row = blockIdx.x;
    const int tid = threadIdx.x;
    const float* xr = x + (size_t)row * BP_D;

    float v0 = xr[tid];
    float v1 = xr[tid + 256];
    float s  = v0 + v1;
    #pragma unroll
    for (int o = 16; o > 0; o >>= 1) s += __shfl_xor(s, o, 32);
    if ((tid & 31) == 0) red[tid >> 5] = s;
    __syncthreads();
    float tot = 0.f;
    #pragma unroll
    for (int i = 0; i < 8; ++i) tot += red[i];
    const float mu = tot * (1.0f / (float)BP_D);

    const float d0 = v0 - mu, d1 = v1 - mu;
    s = d0 * d0 + d1 * d1;
    #pragma unroll
    for (int o = 16; o > 0; o >>= 1) s += __shfl_xor(s, o, 32);
    __syncthreads();
    if ((tid & 31) == 0) red[tid >> 5] = s;
    __syncthreads();
    tot = 0.f;
    #pragma unroll
    for (int i = 0; i < 8; ++i) tot += red[i];
    const float rstd = rsqrtf(tot * (1.0f / (float)BP_D) + 1e-5f);

    y[(size_t)row * BP_D + tid]       = (__bf16)(d0 * rstd * g[tid]       + b[tid]);
    y[(size_t)row * BP_D + tid + 256] = (__bf16)(d1 * rstd * g[tid + 256] + b[tid + 256]);
}

// ---------------------------------------------------------------------------
// TDM: issue one 2D tile load (tile_w x tile_h elements, 2B each) into LDS.
// Descriptor per CDNA5 ISA. Padding: +4 DWORDs
// every 32 DWORDs -> LDS row pitch 144B (RS=72 elements).
// ---------------------------------------------------------------------------
__device__ __forceinline__ void tdm_load_tile(const __bf16* gptr, unsigned lds_byte_off,
                                              unsigned tensor_w, unsigned tensor_h,
                                              unsigned tile_w, unsigned tile_h,
                                              unsigned row_stride)
{
#if HAVE_TDM
    const unsigned long long ga = (unsigned long long)(uintptr_t)gptr;
    u32x4 g0;
    g0[0] = 1u;                                            // count=1 (valid user D#)
    g0[1] = lds_byte_off;                                  // lds_addr
    g0[2] = (unsigned)(ga & 0xFFFFFFFFu);                  // global_addr[31:0]
    g0[3] = (unsigned)((ga >> 32) & 0x01FFFFFFu) | (2u << 30); // global_addr[56:32], type=2
    i32x8 g1;
    g1[0] = (int)((1u << 16)        // data_size = 1 -> 2 bytes
                | (1u << 20)        // pad_enable
                | (4u << 22)        // pad_interval: 2^(4+1)=32 DWORDs
                | (3u << 25));      // pad_amount: 3 -> 4 DWORDs (16B)
    g1[1] = (int)((tensor_w & 0xFFFFu) << 16);                       // tensor_dim0[15:0]
    g1[2] = (int)((tensor_w >> 16) & 0xFFFFu) | (int)((tensor_h & 0xFFFFu) << 16);
    g1[3] = (int)((tensor_h >> 16) & 0xFFFFu) | (int)(tile_w << 16); // tile_dim0
    g1[4] = (int)(tile_h & 0xFFFFu);                                 // tile_dim1 (tile_dim2=0)
    g1[5] = (int)row_stride;                                         // tensor_dim0_stride lo
    g1[6] = 0;                                                       // stride hi / dim1_stride lo
    g1[7] = 0;
    const i32x4 z4 = {0, 0, 0, 0};
#if defined(__clang_major__) && (__clang_major__ >= 23)
    const i32x8 z8 = {0, 0, 0, 0, 0, 0, 0, 0};
    __builtin_amdgcn_tensor_load_to_lds(g0, g1, z4, z4, z8, 0);
#else
    __builtin_amdgcn_tensor_load_to_lds(g0, g1, z4, z4, 0);
#endif
#else
    (void)gptr; (void)lds_byte_off; (void)tensor_w; (void)tensor_h;
    (void)tile_w; (void)tile_h; (void)row_stride;
#endif
}

// Fallback cooperative copy (used only when TDM builtin is unavailable).
__device__ __forceinline__ void stage_copy_fallback(const __bf16* __restrict__ g,
                                                    __bf16* __restrict__ lds, int K)
{
    for (int idx = threadIdx.x; idx < TM * (TK / 8); idx += 256) {
        const int row = idx >> 3;
        const int c8  = (idx & 7) * 8;
        *(bf16x8*)(lds + row * RS + c8) = *(const bf16x8*)(g + (size_t)row * K + c8);
    }
}

// Build a v16bf fragment from two 16-byte LDS reads.
__device__ __forceinline__ v16bf load_frag2(const __bf16* p, int second_off)
{
    const bf16x8 lo = *(const bf16x8*)p;
    const bf16x8 hi = *(const bf16x8*)(p + second_off);
    return __builtin_shufflevector(lo, hi, 0, 1, 2, 3, 4, 5, 6, 7,
                                           8, 9, 10, 11, 12, 13, 14, 15);
}

// ---------------------------------------------------------------------------
// GEMM: C[m,n] = sum_k A[m,k] * W[n,k] (+bias, epilogue).  A:[M,K] bf16,
// W:[N,K] bf16, both row-major. 128x128 block tile, TDM double-buffered
// 64-deep K slabs in LDS, 8 waves each computing a 64x32 sub-tile
// (4x2 accumulators, 16 WMMAs per slab).
//   EPI 0: +bias            -> bf16 out
//   EPI 1: +bias +resid     -> f32 out
//   EPI 2: +bias, exact GELU-> bf16 out
// ---------------------------------------------------------------------------
template<int EPI>
__global__ __launch_bounds__(256)
void gemm_tdm_wmma(const __bf16* __restrict__ A, const __bf16* __restrict__ W,
                   const float* __restrict__ bias, const float* __restrict__ resid,
                   void* __restrict__ out, int M, int N, int K)
{
    __shared__ __align__(16) __bf16 smem[2][2][TM * RS];   // [buf][A/B] 72KB total

    const int lane = threadIdx.x & 31;
    const int wave = threadIdx.x >> 5;
    const int half = lane >> 4;
    const int r    = lane & 15;
    const int wm   = wave >> 2;        // 0..1 -> 64 rows each
    const int wn   = wave & 3;         // 0..3 -> 32 cols each

    const int nbN = N / TN;
    const int bm  = blockIdx.x / nbN;
    const int bn  = blockIdx.x - bm * nbN;

    const __bf16* gA = A + (size_t)bm * TM * K;
    const __bf16* gW = W + (size_t)bn * TN * K;

    const int KS = K / TK;

    // stage 0 prefetch
#if HAVE_TDM
    if (wave == 0) {
        tdm_load_tile(gA, (unsigned)((char*)&smem[0][0][0] - (char*)&smem[0][0][0]) + 0u,
                      (unsigned)K, (unsigned)TM, TK, TM, (unsigned)K);
        tdm_load_tile(gW, (unsigned)(TM * RS * 2),
                      (unsigned)K, (unsigned)TN, TK, TN, (unsigned)K);
        __builtin_amdgcn_s_wait_tensorcnt(0);
    }
#else
    stage_copy_fallback(gA, &smem[0][0][0], K);
    stage_copy_fallback(gW, &smem[0][1][0], K);
#endif
    __syncthreads();

    v8f acc[4][2] = {};

    for (int ks = 0; ks < KS; ++ks) {
        const int cur = ks & 1;

        // kick off next slab into the other buffer (overlaps with compute)
        if (ks + 1 < KS) {
#if HAVE_TDM
            if (wave == 0) {
                const unsigned boff = (unsigned)((cur ^ 1) * 2 * TM * RS * 2);
                tdm_load_tile(gA + (ks + 1) * TK, boff,
                              (unsigned)K, (unsigned)TM, TK, TM, (unsigned)K);
                tdm_load_tile(gW + (ks + 1) * TK, boff + (unsigned)(TM * RS * 2),
                              (unsigned)K, (unsigned)TN, TK, TN, (unsigned)K);
            }
#else
            stage_copy_fallback(gA + (ks + 1) * TK, &smem[cur ^ 1][0][0], K);
            stage_copy_fallback(gW + (ks + 1) * TK, &smem[cur ^ 1][1][0], K);
#endif
        }

        const __bf16* As = &smem[cur][0][0];
        const __bf16* Ws = &smem[cur][1][0];

        #pragma unroll
        for (int kk = 0; kk < TK; kk += 32) {
            v16bf af[4], bfr[2];
            #pragma unroll
            for (int mt = 0; mt < 4; ++mt) {
                // A frag: lane row = wm*64+mt*16+r; K chunks (kk+8*half, +16)
                const __bf16* p = As + (wm * 64 + mt * 16 + r) * RS + kk + 8 * half;
                af[mt] = load_frag2(p, 16);
            }
            #pragma unroll
            for (int nt = 0; nt < 2; ++nt) {
                // B frag: lane col = wn*32+nt*16+r; K chunks (kk+8*half, +16)
                const __bf16* p = Ws + (wn * 32 + nt * 16 + r) * RS + kk + 8 * half;
                bfr[nt] = load_frag2(p, 16);
            }
            #pragma unroll
            for (int mt = 0; mt < 4; ++mt)
                #pragma unroll
                for (int nt = 0; nt < 2; ++nt)
                    acc[mt][nt] = __builtin_amdgcn_wmma_f32_16x16x32_bf16(
                        false, af[mt], false, bfr[nt], (short)0, acc[mt][nt],
                        false, false);
        }

        __syncthreads();                     // all waves done with buf[cur]
#if HAVE_TDM
        if (wave == 0) __builtin_amdgcn_s_wait_tensorcnt(0);
#endif
        __syncthreads();                     // next buffer ready for everyone
    }

    // Epilogue. C/D layout: VGPR i -> M = i + 8*half, N = r within a 16x16 tile.
    const int m_base = bm * TM + wm * 64;
    const int n_base = bn * TN + wn * 32;
    #pragma unroll
    for (int nt = 0; nt < 2; ++nt) {
        const int n  = n_base + nt * 16 + r;
        const float bnv = bias[n];
        #pragma unroll
        for (int mt = 0; mt < 4; ++mt) {
            #pragma unroll
            for (int i = 0; i < 8; ++i) {
                const int m = m_base + mt * 16 + i + 8 * half;
                float v = acc[mt][nt][i] + bnv;
                if (EPI == 2)
                    v = 0.5f * v * (1.0f + erff(v * 0.70710678118654752f));
                if (EPI == 1) {
                    v += resid[(size_t)m * N + n];
                    ((float*)out)[(size_t)m * N + n] = v;
                } else {
                    ((__bf16*)out)[(size_t)m * N + n] = (__bf16)v;
                }
            }
        }
    }
}

// ---------------------------------------------------------------------------
// Attention: one 256-thread block per (b,t). Q,K,V bf16; fp32 math; bf16 out.
// ---------------------------------------------------------------------------
__global__ __launch_bounds__(256)
void attention_kernel(const __bf16* __restrict__ Q, const __bf16* __restrict__ Kp,
                      const __bf16* __restrict__ Vp, __bf16* __restrict__ att)
{
    __shared__ float qs[BP_D];
    __shared__ float sc[BP_H];

    const int bt   = blockIdx.x;               // b*T + t
    const int t    = bt & (BP_T - 1);
    const int tid  = threadIdx.x;
    const int lane = tid & 31;
    const int wave = tid >> 5;

    const __bf16* qrow = Q + (size_t)bt * BP_D;
    qs[tid]       = (float)qrow[tid];
    qs[tid + 256] = (float)qrow[tid + 256];
    __syncthreads();

    const __bf16* Kbase = Kp + (size_t)t * BP_H * BP_D;
    const float   scale = 0.04419417382415922f;     // 1/sqrt(512)

    #pragma unroll
    for (int hh = 0; hh < 8; ++hh) {
        const int h = wave * 8 + hh;
        const __bf16* Kr = Kbase + (size_t)h * BP_D;
        float p = 0.f;
        #pragma unroll 4
        for (int i = lane; i < BP_D; i += 32) p += qs[i] * (float)Kr[i];
        #pragma unroll
        for (int o = 16; o > 0; o >>= 1) p += __shfl_xor(p, o, 32);
        if (lane == 0) {
            const float dec = powf(0.95f, (float)(BP_H - 1 - h));
            sc[h] = p * scale + logf(dec + 1e-10f);
        }
    }
    __syncthreads();

    float mx = -3.0e38f;
    #pragma unroll
    for (int h = 0; h < BP_H; ++h) mx = fmaxf(mx, sc[h]);
    __syncthreads();
    if (tid < BP_H) sc[tid] = expf(sc[tid] - mx);
    __syncthreads();
    float sum = 0.f;
    #pragma unroll
    for (int h = 0; h < BP_H; ++h) sum += sc[h];
    const float inv = 1.0f / sum;

    const __bf16* Vbase = Vp + (size_t)t * BP_H * BP_D;
    float acc0 = 0.f, acc1 = 0.f;
    for (int h = 0; h < BP_H; ++h) {
        const float w = sc[h];
        acc0 += w * (float)Vbase[(size_t)h * BP_D + tid];
        acc1 += w * (float)Vbase[(size_t)h * BP_D + tid + 256];
    }
    att[(size_t)bt * BP_D + tid]       = (__bf16)(acc0 * inv);
    att[(size_t)bt * BP_D + tid + 256] = (__bf16)(acc1 * inv);
}

// ---------------------------------------------------------------------------
// Host-side launch
// ---------------------------------------------------------------------------
extern "C" void kernel_launch(void* const* d_in, const int* in_sizes, int n_in,
                              void* d_out, int out_size, void* d_ws, size_t ws_size,
                              hipStream_t stream)
{
    const float* x    = (const float*)d_in[0];
    const float* hist = (const float*)d_in[1];
    const float* Wq   = (const float*)d_in[2];
    const float* bq   = (const float*)d_in[3];
    const float* Wk   = (const float*)d_in[4];
    const float* bk   = (const float*)d_in[5];
    const float* Wv   = (const float*)d_in[6];
    const float* bv   = (const float*)d_in[7];
    const float* Wo   = (const float*)d_in[8];
    const float* bo   = (const float*)d_in[9];
    const float* W1   = (const float*)d_in[10];
    const float* b1   = (const float*)d_in[11];
    const float* W2   = (const float*)d_in[12];
    const float* b2   = (const float*)d_in[13];
    const float* g1   = (const float*)d_in[14];
    const float* be1  = (const float*)d_in[15];
    const float* g2   = (const float*)d_in[16];
    const float* be2  = (const float*)d_in[17];
    float* out = (float*)d_out;

    const size_t BT = (size_t)BP_B * BP_T;     // 16384 token rows
    const size_t TH = (size_t)BP_T * BP_H;     // 131072 history rows
    const size_t D  = BP_D;
    const size_t DD = D * D;

    // workspace carve-up (bf16 activations, x2 fp32)
    char* w = (char*)d_ws;
    __bf16* histb = (__bf16*)w;                 w += TH * D * 2;
    __bf16* Wqb   = (__bf16*)w;                 w += DD * 2;
    __bf16* Wkb   = (__bf16*)w;                 w += DD * 2;
    __bf16* Wvb   = (__bf16*)w;                 w += DD * 2;
    __bf16* Wob   = (__bf16*)w;                 w += DD * 2;
    __bf16* W1b   = (__bf16*)w;                 w += 2 * DD * 2;
    __bf16* W2b   = (__bf16*)w;                 w += 2 * DD * 2;
    __bf16* xn    = (__bf16*)w;                 w += BT * D * 2;
    __bf16* Qb    = (__bf16*)w;                 w += BT * D * 2;
    __bf16* Kb    = (__bf16*)w;                 w += TH * D * 2;
    __bf16* Vb    = (__bf16*)w;                 w += TH * D * 2;
    __bf16* at    = (__bf16*)w;                 w += BT * D * 2;
    float*  x2    = (float*)w;                  w += BT * D * 4;
    __bf16* h2    = (__bf16*)w;                 w += BT * D * 2;
    __bf16* f1    = (__bf16*)w;                 w += BT * 2 * D * 2;

    // one-time fp32 -> bf16 conversions
    cvt_kernel<<<(int)(TH * D / 1024), 256, 0, stream>>>(hist, histb, (int)(TH * D / 4));
    cvt_kernel<<<(int)(DD / 1024), 256, 0, stream>>>(Wq, Wqb, (int)(DD / 4));
    cvt_kernel<<<(int)(DD / 1024), 256, 0, stream>>>(Wk, Wkb, (int)(DD / 4));
    cvt_kernel<<<(int)(DD / 1024), 256, 0, stream>>>(Wv, Wvb, (int)(DD / 4));
    cvt_kernel<<<(int)(DD / 1024), 256, 0, stream>>>(Wo, Wob, (int)(DD / 4));
    cvt_kernel<<<(int)(2 * DD / 1024), 256, 0, stream>>>(W1, W1b, (int)(2 * DD / 4));
    cvt_kernel<<<(int)(2 * DD / 1024), 256, 0, stream>>>(W2, W2b, (int)(2 * DD / 4));

    const int gQ  = (int)((BT / TM) * (D / TN));        // 512
    const int gKV = (int)((TH / TM) * (D / TN));        // 4096
    const int gF1 = (int)((BT / TM) * (2 * D / TN));    // 1024

    // 1) xn = LN(x)
    ln_kernel<<<(int)BT, 256, 0, stream>>>(x, g1, be1, xn);
    // 2-4) Q/K/V projections
    gemm_tdm_wmma<0><<<gQ, 256, 0, stream>>>(xn, Wqb, bq, nullptr, Qb,
                                             (int)BT, (int)D, (int)D);
    gemm_tdm_wmma<0><<<gKV, 256, 0, stream>>>(histb, Wkb, bk, nullptr, Kb,
                                              (int)TH, (int)D, (int)D);
    gemm_tdm_wmma<0><<<gKV, 256, 0, stream>>>(histb, Wvb, bv, nullptr, Vb,
                                              (int)TH, (int)D, (int)D);
    // 5) decayed softmax attention
    attention_kernel<<<(int)BT, 256, 0, stream>>>(Qb, Kb, Vb, at);
    // 6) x2 = x + attended @ Wo.T + bo
    gemm_tdm_wmma<1><<<gQ, 256, 0, stream>>>(at, Wob, bo, x, x2,
                                             (int)BT, (int)D, (int)D);
    // 7) h2 = LN(x2)
    ln_kernel<<<(int)BT, 256, 0, stream>>>(x2, g2, be2, h2);
    // 8) f1 = gelu(h2 @ W1.T + b1)
    gemm_tdm_wmma<2><<<gF1, 256, 0, stream>>>(h2, W1b, b1, nullptr, f1,
                                              (int)BT, (int)(2 * D), (int)D);
    // 9) out = x2 + f1 @ W2.T + b2
    gemm_tdm_wmma<1><<<gQ, 256, 0, stream>>>(f1, W2b, b2, x2, out,
                                             (int)BT, (int)D, (int)(2 * D));
}
